// WarpingLayers_2697239462394
// MI455X (gfx1250) — matrix operand order repaired and checked by reading, not compiled, not run on hardware
//
#include <hip/hip_runtime.h>
#include <math.h>

typedef _Float16 v16h __attribute__((ext_vector_type(16)));
typedef float    v8f  __attribute__((ext_vector_type(8)));

#define H_IMG  64
#define W_IMG  256
#define NIMG   16384          // H*W
#define BNTOT  65536          // B*H*W
#define NWAVES 8

// ---------------------------------------------------------------------------
// WMMA fragment loaders (CDNA5 v_wmma_f32_16x16x32_f16, wave32)
// A (16x32 f16): lane L holds row m = L&15.
//   a[e] (e=0..7)  -> K = e      + 8*(L>>4)
//   a[e] (e=8..15) -> K = e + 8  + 8*(L>>4)
// B (32x16 f16): lane L holds col n = L&15, same K mapping as A
// C/D (16x16 f32): c[r] -> row m = r + 8*(L>>4), col n = L&15
// ---------------------------------------------------------------------------
__device__ __forceinline__ v16h load_a_frag(const _Float16* A, int lda, int m0,
                                            int k0, int lane) {
  const _Float16* p = A + (m0 + (lane & 15)) * lda + k0 + 8 * (lane >> 4);
  v16h r;
#pragma unroll
  for (int i = 0; i < 8; ++i) { r[i] = p[i]; r[i + 8] = p[i + 16]; }
  return r;
}

__device__ __forceinline__ v16h load_b_frag(const _Float16* WT, int kp, int n0,
                                            int k0, int lane) {
  const _Float16* p = WT + (n0 + (lane & 15)) * kp + k0 + 8 * (lane >> 4);
  v16h r;
#pragma unroll
  for (int i = 0; i < 8; ++i) { r[i] = p[i]; r[i + 8] = p[i + 16]; }
  return r;
}

// D = relu(A[TM x Kp] * W + bias).  A: f16 LDS row-major (lda).
// WT: f16 global, transposed [Nout][Kp].  Output: f16 (Dh) or f32 (Df), ldd/dcol.
__device__ __forceinline__ void gemm_relu(
    const _Float16* __restrict__ A, int lda, int Kp,
    const _Float16* __restrict__ WT, const float* __restrict__ bias,
    _Float16* __restrict__ Dh, float* __restrict__ Df, int ldd, int dcol,
    int Tm, int Nout)
{
  const int wave = threadIdx.x >> 5;
  const int lane = threadIdx.x & 31;
  const int Tn = Nout >> 4;
  const int total = Tm * Tn;
  for (int tile = wave; tile < total; tile += NWAVES) {   // wave-uniform
    const int m0 = (tile / Tn) << 4;
    const int n0 = (tile % Tn) << 4;
    v8f acc = {0.f, 0.f, 0.f, 0.f, 0.f, 0.f, 0.f, 0.f};
    for (int k0 = 0; k0 < Kp; k0 += 32) {
      v16h af = load_a_frag(A, lda, m0, k0, lane);
      v16h bf = load_b_frag(WT, Kp, n0, k0, lane);
      acc = __builtin_amdgcn_wmma_f32_16x16x32_f16(
          false, af, false, bf, (short)0, acc, false, false);
    }
    const int   n  = n0 + (lane & 15);
    const float bv = bias[n];
    const int   mr = m0 + ((lane >> 4) << 3);
#pragma unroll
    for (int r = 0; r < 8; ++r) {
      float v = acc[r] + bv;
      v = v > 0.f ? v : 0.f;
      if (Df) Df[(mr + r) * ldd + dcol + n] = v;
      else    Dh[(mr + r) * ldd + dcol + n] = (_Float16)v;
    }
  }
}

// ---------------------------------------------------------------------------
// Weight prep: f32 [K][N] -> f16 transposed [N][Kp], zero-padded rows k>=K.
// ---------------------------------------------------------------------------
__global__ __launch_bounds__(256) void prep_w_kernel(
    const float* __restrict__ w, _Float16* __restrict__ wt,
    int K, int Nc, int Kp)
{
  int i = blockIdx.x * 256 + threadIdx.x;
  if (i >= Nc * Kp) return;
  int n = i / Kp, k = i - n * Kp;
  wt[i] = (k < K) ? (_Float16)w[k * Nc + n] : (_Float16)0.0f;
}

// ---------------------------------------------------------------------------
// Windowed kNN select (emulates fused_conv_select_k).
// ---------------------------------------------------------------------------
template<int KH, int KW, int K>
__global__ __launch_bounds__(256) void select_k_kernel(
    const float* __restrict__ xq, const float* __restrict__ xr,
    float distThr, int* __restrict__ idx, float* __restrict__ mask)
{
  int g = blockIdx.x * 256 + threadIdx.x;
  if (g >= BNTOT) return;
  int b = g / NIMG;
  int n = g - b * NIMG;
  int h = n >> 8;           // W = 256
  int w = n & 255;
  const float* q = xq + (size_t)g * 3;
  float qx = q[0], qy = q[1], qz = q[2];
  float bd[K]; int bi[K];
#pragma unroll
  for (int j = 0; j < K; ++j) { bd[j] = 3.0e38f; bi[j] = 0; }
  for (int dh = -(KH / 2); dh <= KH / 2; ++dh) {
    int h2 = h + dh;
    if (h2 < 0 || h2 >= H_IMG) continue;
    for (int dw = -(KW / 2); dw <= KW / 2; ++dw) {
      int w2 = w + dw;
      if (w2 < 0 || w2 >= W_IMG) continue;
      int f = (h2 << 8) + w2;
      const float* r = xr + (size_t)(b * NIMG + f) * 3;
      float dx = r[0] - qx, dy = r[1] - qy, dz = r[2] - qz;
      float d2 = dx * dx + dy * dy + dz * dz;
      if (d2 >= distThr) continue;
      if (d2 < bd[K - 1]) {
        bd[K - 1] = d2; bi[K - 1] = f;
#pragma unroll
        for (int j = K - 1; j > 0; --j) {
          if (bd[j] < bd[j - 1]) {
            float td = bd[j]; bd[j] = bd[j - 1]; bd[j - 1] = td;
            int   ti = bi[j]; bi[j] = bi[j - 1]; bi[j - 1] = ti;
          }
        }
      }
    }
  }
#pragma unroll
  for (int j = 0; j < K; ++j) {
    bool ok = bd[j] < 1.0e38f;
    idx [g * K + j] = ok ? bi[j] : 0;
    mask[g * K + j] = ok ? 1.0f : 0.0f;
  }
}

// ---------------------------------------------------------------------------
// Stage 1 fused: 16 pixels/block -> 96 rows (kq=6). All 6 GEMMs + softmax
// kept in LDS; only pi_feat1_new [BN][64] f32 leaves the kernel.
// LDS (92160 B): X1 f16[96][160] @0 | H1 f16[96][128] @30720 |
//   H2 f16[96][64] @55296 | EF f16[96][128] @67584 | H3 aliases X1 |
//   Att f32[96][64] aliases H1.
// ---------------------------------------------------------------------------
__global__ __launch_bounds__(256) void stage1_fused(
    const float* __restrict__ xyz1, const float* __restrict__ xyz2,
    const float* __restrict__ pts1, const float* __restrict__ pts2,
    const int*   __restrict__ idxq, const float* __restrict__ maskq,
    const _Float16* __restrict__ wt0, const float* __restrict__ b0,
    const _Float16* __restrict__ wt1, const float* __restrict__ b1,
    const _Float16* __restrict__ wt2, const float* __restrict__ b2,
    const _Float16* __restrict__ wtpe, const float* __restrict__ bpe,
    const _Float16* __restrict__ wt3, const float* __restrict__ b3,
    const _Float16* __restrict__ wt4, const float* __restrict__ b4,
    float* __restrict__ piF32)
{
  __shared__ __align__(16) char smem[92160];
  _Float16* sX1 = (_Float16*)(smem);           // [96][160]
  _Float16* sH1 = (_Float16*)(smem + 30720);   // [96][128]
  _Float16* sH2 = (_Float16*)(smem + 55296);   // [96][64]
  _Float16* sEF = (_Float16*)(smem + 67584);   // [96][128] enc|feat
  _Float16* sH3 = sX1;                         // alias (X1 dead after L1/ENC)
  float*    sAtt = (float*)(smem + 30720);     // alias H1 (dead after L2)

  const int P0 = blockIdx.x * 16;              // pixel base (over B*N)
  const int t  = threadIdx.x;

  // ---- assemble xyz_cat (10ch) + zero pad cols 138..159 ----
  if (t < 96) {
    int pl = t / 6, j = t - pl * 6;
    int g  = P0 + pl;
    int b  = g / NIMG;
    int gi = b * NIMG + idxq[g * 6 + j];
    float mk = maskq[g * 6 + j];
    float px = xyz1[g * 3], py = xyz1[g * 3 + 1], pz = xyz1[g * 3 + 2];
    float qx = xyz2[gi * 3] * mk, qy = xyz2[gi * 3 + 1] * mk, qz = xyz2[gi * 3 + 2] * mk;
    float dx = qx - px, dy = qy - py, dz = qz - pz;
    float e  = sqrtf(dx * dx + dy * dy + dz * dz + 1e-20f);
    _Float16* row = sX1 + t * 160;
    row[0] = (_Float16)px; row[1] = (_Float16)py; row[2] = (_Float16)pz;
    row[3] = (_Float16)qx; row[4] = (_Float16)qy; row[5] = (_Float16)qz;
    row[6] = (_Float16)dx; row[7] = (_Float16)dy; row[8] = (_Float16)dz;
    row[9] = (_Float16)e;
#pragma unroll
    for (int c = 138; c < 160; ++c) row[c] = (_Float16)0.f;
  }
  // ---- assemble pi_pts | qi_pts (128ch) ----
  for (int i = t; i < 96 * 128; i += 256) {
    int row = i >> 7, c = i & 127;
    int pl = row / 6, j = row - pl * 6;
    int g  = P0 + pl;
    float v;
    if (c < 64) {
      v = pts1[g * 64 + c];
    } else {
      int b  = g / NIMG;
      int gi = b * NIMG + idxq[g * 6 + j];
      v = pts2[gi * 64 + (c - 64)] * maskq[g * 6 + j];
    }
    sX1[row * 160 + 10 + c] = (_Float16)v;
  }
  __syncthreads();

  gemm_relu(sX1, 160, 160, wt0,  b0,  sH1, nullptr, 128, 0,  6, 128); // mlp1[0]
  gemm_relu(sX1, 160, 32,  wtpe, bpe, sEF, nullptr, 128, 0,  6, 64);  // pi_encoding
  __syncthreads();
  gemm_relu(sH1, 128, 128, wt1,  b1,  sH2, nullptr, 64,  0,  6, 64);  // mlp1[1]
  __syncthreads();
  gemm_relu(sH2, 64,  64,  wt2,  b2,  sEF, nullptr, 128, 64, 6, 64);  // mlp1[2]
  __syncthreads();
  gemm_relu(sEF, 128, 128, wt3,  b3,  sH3, nullptr, 128, 0,  6, 128); // mlp2[0]
  __syncthreads();
  gemm_relu(sH3, 128, 128, wt4,  b4,  nullptr, sAtt, 64, 0,  6, 64);  // mlp2[1]
  __syncthreads();

  // ---- masked softmax over kq=6 + weighted sum -> pi_feat1_new ----
  for (int i = t; i < 16 * 64; i += 256) {
    int pl = i >> 6, d = i & 63;
    int g  = P0 + pl;
    float x[6];
    float mx = -3.0e38f;
#pragma unroll
    for (int j = 0; j < 6; ++j) {
      float mk = maskq[g * 6 + j];
      float v  = sAtt[(pl * 6 + j) * 64 + d];
      x[j] = (mk > 0.5f) ? v : -1.0e10f;
      mx = fmaxf(mx, x[j]);
    }
    float s = 0.f, acc = 0.f;
#pragma unroll
    for (int j = 0; j < 6; ++j) {
      float e = __expf(x[j] - mx);
      s   += e;
      acc += e * (float)sEF[(pl * 6 + j) * 128 + 64 + d];
    }
    piF32[g * 64 + d] = acc / s;
  }
}

// ---------------------------------------------------------------------------
// Stage 2 fused: 16 pixels/block -> 64 rows (k=4).
// LDS (77824 B): Xe f16[64][32] @0 | C2 f16[64][192] @4096 |
//   H4 f16[64][128] @28672 | WPl f32[64][64] @45056 | G2 f32[64][64] @61440
// ---------------------------------------------------------------------------
__global__ __launch_bounds__(256) void stage2_fused(
    const float* __restrict__ xyz1, const float* __restrict__ pts1,
    const float* __restrict__ piF32,
    const int*   __restrict__ idxp, const float* __restrict__ maskp,
    const _Float16* __restrict__ wtpc, const float* __restrict__ bpc,
    const _Float16* __restrict__ wtn0, const float* __restrict__ bn0,
    const _Float16* __restrict__ wtn1, const float* __restrict__ bn1,
    float* __restrict__ out)
{
  __shared__ __align__(16) char smem[77824];
  _Float16* sXe  = (_Float16*)(smem);           // [64][32]
  _Float16* sC2  = (_Float16*)(smem + 4096);    // [64][192] enc|pn|pg
  _Float16* sH4  = (_Float16*)(smem + 28672);   // [64][128]
  float*    sWP  = (float*)   (smem + 45056);   // [64][64]
  float*    sG2  = (float*)   (smem + 61440);   // [64][64] masked pc_pts_g f32

  const int P0 = blockIdx.x * 16;
  const int t  = threadIdx.x;

  // ---- xyz_cat2 (10ch, pad to 32) ----
  if (t < 64) {
    int pl = t >> 2, j = t & 3;
    int g  = P0 + pl;
    int b  = g / NIMG;
    int gi = b * NIMG + idxp[g * 4 + j];
    float mk = maskp[g * 4 + j];
    float px = xyz1[g * 3], py = xyz1[g * 3 + 1], pz = xyz1[g * 3 + 2];
    float qx = xyz1[gi * 3] * mk, qy = xyz1[gi * 3 + 1] * mk, qz = xyz1[gi * 3 + 2] * mk;
    float dx = qx - px, dy = qy - py, dz = qz - pz;
    float e  = sqrtf(dx * dx + dy * dy + dz * dz + 1e-20f);
    _Float16* row = sXe + t * 32;
    row[0] = (_Float16)px; row[1] = (_Float16)py; row[2] = (_Float16)pz;
    row[3] = (_Float16)qx; row[4] = (_Float16)qy; row[5] = (_Float16)qz;
    row[6] = (_Float16)dx; row[7] = (_Float16)dy; row[8] = (_Float16)dz;
    row[9] = (_Float16)e;
#pragma unroll
    for (int c = 10; c < 32; ++c) row[c] = (_Float16)0.f;
  }
  // ---- pc_pts_n*mask (cols 64..127) and pc_pts_g*mask (cols 128..191) ----
  for (int i = t; i < 64 * 128; i += 256) {
    int row = i >> 7, c = i & 127;
    int pl = row >> 2, j = row & 3;
    int g  = P0 + pl;
    float mk = maskp[g * 4 + j];
    if (c < 64) {
      sC2[row * 192 + 64 + c] = (_Float16)(pts1[g * 64 + c] * mk);
    } else {
      int b  = g / NIMG;
      int gi = b * NIMG + idxp[g * 4 + j];
      float v = piF32[gi * 64 + (c - 64)] * mk;
      sC2[row * 192 + 128 + (c - 64)] = (_Float16)v;
      sG2[row * 64 + (c - 64)] = v;
    }
  }
  __syncthreads();
  gemm_relu(sXe, 32, 32, wtpc, bpc, sC2, nullptr, 192, 0, 4, 64);   // pc_encoding
  __syncthreads();
  // apply mask to enc2 columns (whole concat is masked in the reference)
  for (int i = t; i < 64 * 64; i += 256) {
    int row = i >> 6, c = i & 63;
    int pl = row >> 2, j = row & 3;
    float mk = maskp[(P0 + pl) * 4 + j];
    sC2[row * 192 + c] = (_Float16)((float)sC2[row * 192 + c] * mk);
  }
  __syncthreads();
  gemm_relu(sC2, 192, 192, wtn0, bn0, sH4, nullptr, 128, 0, 4, 128); // mlp2_new[0]
  __syncthreads();
  gemm_relu(sH4, 128, 128, wtn1, bn1, nullptr, sWP, 64, 0, 4, 64);   // mlp2_new[1]
  __syncthreads();

  // ---- masked softmax over k=4 + weighted sum of pc_pts_g ----
  for (int i = t; i < 16 * 64; i += 256) {
    int pl = i >> 6, d = i & 63;
    int g  = P0 + pl;
    float x[4];
    float mx = -3.0e38f;
#pragma unroll
    for (int j = 0; j < 4; ++j) {
      float mk = maskp[g * 4 + j];
      float v  = sWP[(pl * 4 + j) * 64 + d];
      x[j] = (mk > 0.5f) ? v : -1.0e10f;
      mx = fmaxf(mx, x[j]);
    }
    float s = 0.f, acc = 0.f;
#pragma unroll
    for (int j = 0; j < 4; ++j) {
      float e = __expf(x[j] - mx);
      s   += e;
      acc += e * sG2[(pl * 4 + j) * 64 + d];
    }
    out[g * 64 + d] = acc / s;
  }
}

// ---------------------------------------------------------------------------
// Host launcher
// ---------------------------------------------------------------------------
extern "C" void kernel_launch(void* const* d_in, const int* in_sizes, int n_in,
                              void* d_out, int out_size, void* d_ws, size_t ws_size,
                              hipStream_t stream)
{
  (void)in_sizes; (void)n_in; (void)out_size; (void)ws_size;

  const float* xyz1 = (const float*)d_in[0];
  const float* xyz2 = (const float*)d_in[1];
  const float* pts1 = (const float*)d_in[2];
  const float* pts2 = (const float*)d_in[3];
  const float* m1w0 = (const float*)d_in[4];  const float* m1b0 = (const float*)d_in[5];
  const float* m1w1 = (const float*)d_in[6];  const float* m1b1 = (const float*)d_in[7];
  const float* m1w2 = (const float*)d_in[8];  const float* m1b2 = (const float*)d_in[9];
  const float* pew  = (const float*)d_in[10]; const float* peb  = (const float*)d_in[11];
  const float* m2w0 = (const float*)d_in[12]; const float* m2b0 = (const float*)d_in[13];
  const float* m2w1 = (const float*)d_in[14]; const float* m2b1 = (const float*)d_in[15];
  const float* pcw  = (const float*)d_in[16]; const float* pcb  = (const float*)d_in[17];
  const float* n0w  = (const float*)d_in[18]; const float* n0b  = (const float*)d_in[19];
  const float* n1w  = (const float*)d_in[20]; const float* n1b  = (const float*)d_in[21];

  char* ws = (char*)d_ws;
  _Float16* wt0   = (_Float16*)(ws + 0);        // [128][160]
  _Float16* wt1   = (_Float16*)(ws + 40960);    // [64][128]
  _Float16* wt2   = (_Float16*)(ws + 57344);    // [64][64]
  _Float16* wtpe  = (_Float16*)(ws + 65536);    // [64][32]
  _Float16* wt3   = (_Float16*)(ws + 69632);    // [128][128]
  _Float16* wt4   = (_Float16*)(ws + 102400);   // [64][128]
  _Float16* wtpc  = (_Float16*)(ws + 118784);   // [64][32]
  _Float16* wtn0  = (_Float16*)(ws + 122880);   // [128][192]
  _Float16* wtn1  = (_Float16*)(ws + 172032);   // [64][128]
  int*      idxq  = (int*)     (ws + 188416);   // [BN][6]
  float*    maskq = (float*)   (ws + 1761280);  // [BN][6]
  int*      idxp  = (int*)     (ws + 3334144);  // [BN][4]
  float*    maskp = (float*)   (ws + 4382720);  // [BN][4]
  float*    piF32 = (float*)   (ws + 5431296);  // [BN][64]

  // 1) weight prep (f32 [K][N] -> f16 [N][Kpad])
  struct { const float* w; int K, N, Kp; _Float16* dst; } P[9] = {
    { m1w0, 138, 128, 160, wt0  },
    { m1w1, 128,  64, 128, wt1  },
    { m1w2,  64,  64,  64, wt2  },
    { pew,   10,  64,  32, wtpe },
    { m2w0, 128, 128, 128, wt3  },
    { m2w1, 128,  64, 128, wt4  },
    { pcw,   10,  64,  32, wtpc },
    { n0w,  192, 128, 192, wtn0 },
    { n1w,  128,  64, 128, wtn1 },
  };
  for (int i = 0; i < 9; ++i) {
    int tot = P[i].N * P[i].Kp;
    prep_w_kernel<<<(tot + 255) / 256, 256, 0, stream>>>(
        P[i].w, P[i].dst, P[i].K, P[i].N, P[i].Kp);
  }

  // 2) kNN selects
  select_k_kernel<5, 15, 6><<<BNTOT / 256, 256, 0, stream>>>(
      xyz1, xyz2, 100.0f, idxq, maskq);
  select_k_kernel<3, 3, 4><<<BNTOT / 256, 256, 0, stream>>>(
      xyz1, xyz1, 10.0f, idxp, maskp);

  // 3) stage 1: cross-frame attentive cost volume (fused WMMA MLP chain)
  stage1_fused<<<BNTOT / 16, 256, 0, stream>>>(
      xyz1, xyz2, pts1, pts2, idxq, maskq,
      wt0, m1b0, wt1, m1b1, wt2, m1b2, wtpe, peb, wt3, m2b0, wt4, m2b1,
      piF32);

  // 4) stage 2: self-aggregation (fused WMMA MLP chain) -> d_out
  stage2_fused<<<BNTOT / 16, 256, 0, stream>>>(
      xyz1, pts1, piF32, idxp, maskp,
      wtpc, pcb, wtn0, n0b, wtn1, n1b,
      (float*)d_out);
}
